// StepLocProxy_2662879723812
// MI455X (gfx1250) — matrix operand order repaired and checked by reading, not compiled, not run on hardware
//
#include <hip/hip_runtime.h>
#include <math.h>

// ---- problem constants --------------------------------------------------
#define B_   256
#define N_   256
#define D_   512
#define H_   8
#define E_   256
#define CD_  64
#define NB_  32
#define NP_  16
#define GEO_ 51
#define DH_  64
#define L_   258     // N + 2 cls tokens

typedef __attribute__((ext_vector_type(16))) _Float16 v16h;
typedef __attribute__((ext_vector_type(8)))  float    v8f;

__device__ __forceinline__ float gelu_exact(float x) {
  return 0.5f * x * (1.0f + erff(x * 0.70710678118654752f));
}

// ---- CDNA5 async global->LDS staging (guarded) --------------------------
#if defined(__HIP_DEVICE_COMPILE__) && __has_builtin(__builtin_amdgcn_global_load_async_to_lds_b128)
#define USE_ASYNC_LDS 1
typedef int async_v4i __attribute__((vector_size(16)));
#define ASYNC_CPY_B128(gsrc, ldst)                                          \
  __builtin_amdgcn_global_load_async_to_lds_b128(                           \
      (__attribute__((address_space(1))) async_v4i*)(gsrc),                 \
      (__attribute__((address_space(3))) async_v4i*)(ldst), 0, 0)
#if __has_builtin(__builtin_amdgcn_s_wait_asynccnt)
#define WAIT_ASYNC() __builtin_amdgcn_s_wait_asynccnt(0)
#else
#define WAIT_ASYNC() asm volatile("s_wait_asynccnt 0" ::: "memory")
#endif
#else
#define USE_ASYNC_LDS 0
#endif

// ---- small utility kernels ---------------------------------------------
__global__ void zero_scalar_kernel(float* p) { *p = 0.0f; }

__global__ void f32_to_f16_kernel(const float* __restrict__ src,
                                  _Float16* __restrict__ dst, int n) {
  int i = blockIdx.x * blockDim.x + threadIdx.x;
  if (i < n) dst[i] = (_Float16)src[i];
}

// xin[row, 0:256] = emb[step_ids[row]]; xin[row, 256:320] = coords[row]
__global__ void build_xin_kernel(const int* __restrict__ ids,
                                 const float* __restrict__ coords,
                                 const float* __restrict__ emb,
                                 _Float16* __restrict__ xin) {
  long i = (long)blockIdx.x * blockDim.x + threadIdx.x;   // over 65536*320
  int row = (int)(i / 320);
  int c   = (int)(i - (long)row * 320);
  float v;
  if (c < E_) v = emb[(long)ids[row] * E_ + c];
  else        v = coords[(long)row * CD_ + (c - E_)];
  xin[i] = (_Float16)v;
}

// ---- generic WMMA GEMM:  out = act(A[MxK] @ W[NoutxK]^T + bias) ---------
// A, W f16 row-major. 8 waves; workgroup tile 64(M) x 256(N); K-step 32.
// Per wave: 32x64 of output = 2 A-frags x 4 B-frags = 8 WMMA per K-step.
// Double-buffered LDS tiles; async global->LDS staging when available.
#define GT_M 64
#define GT_N 256
#define TPAD 40   // padded half-stride for LDS tiles (keeps 16B alignment)

__launch_bounds__(256)
__global__ void wmma_gemm_kernel(const _Float16* __restrict__ A,
                                 const _Float16* __restrict__ W,
                                 const float* __restrict__ bias,
                                 float* __restrict__ outF32,
                                 _Float16* __restrict__ outF16,
                                 int M, int K, int Nout, int act) {
  __shared__ __align__(16) _Float16 Atile[2][GT_M * TPAD];
  __shared__ __align__(16) _Float16 Btile[2][GT_N * TPAD];

  const int tid  = threadIdx.x;
  const int lane = tid & 31;
  const int wave = tid >> 5;
  const int wm   = wave >> 2;          // 0..1  (32-row half)
  const int wn   = wave & 3;           // 0..3  (64-col quarter)
  const int h    = lane >> 4;          // lane half
  const int nl   = lane & 15;

  const int mBase = blockIdx.x * GT_M;
  const int nBase = blockIdx.y * GT_N;

  // staging: each thread owns one B row (32 halves) + one 8-half A chunk
  const _Float16* wSrc = W + (size_t)(nBase + tid) * K;
  const int aRow  = tid >> 2;          // 0..63
  const int aPart = tid & 3;           // 0..3
  const _Float16* aSrc = A + (size_t)(mBase + aRow) * K + aPart * 8;

  _Float16* bDst0 = &Btile[0][tid * TPAD];
  _Float16* bDst1 = &Btile[1][tid * TPAD];
  _Float16* aDst0 = &Atile[0][aRow * TPAD + aPart * 8];
  _Float16* aDst1 = &Atile[1][aRow * TPAD + aPart * 8];

  v8f acc[2][4] = {};

  const int ksteps = K >> 5;

  // ---- prologue: stage tile 0 into buffer 0 ----
#if USE_ASYNC_LDS
  ASYNC_CPY_B128(wSrc + 0,  bDst0 + 0);
  ASYNC_CPY_B128(wSrc + 8,  bDst0 + 8);
  ASYNC_CPY_B128(wSrc + 16, bDst0 + 16);
  ASYNC_CPY_B128(wSrc + 24, bDst0 + 24);
  ASYNC_CPY_B128(aSrc, aDst0);
  WAIT_ASYNC();
  __syncthreads();
#else
  {
    const uint4* bs = (const uint4*)wSrc;
    uint4 t0 = bs[0], t1 = bs[1], t2 = bs[2], t3 = bs[3];
    uint4 ta = *(const uint4*)aSrc;
    uint4* bd = (uint4*)bDst0;
    bd[0] = t0; bd[1] = t1; bd[2] = t2; bd[3] = t3;
    *(uint4*)aDst0 = ta;
    __syncthreads();
  }
#endif

  for (int kk = 0; kk < ksteps; ++kk) {
    const int cur = kk & 1;
    const bool hasNext = (kk + 1) < ksteps;
    const int k1 = (kk + 1) << 5;
    _Float16* bN = cur ? bDst0 : bDst1;
    _Float16* aN = cur ? aDst0 : aDst1;

#if USE_ASYNC_LDS
    if (hasNext) {
      ASYNC_CPY_B128(wSrc + k1 + 0,  bN + 0);
      ASYNC_CPY_B128(wSrc + k1 + 8,  bN + 8);
      ASYNC_CPY_B128(wSrc + k1 + 16, bN + 16);
      ASYNC_CPY_B128(wSrc + k1 + 24, bN + 24);
      ASYNC_CPY_B128(aSrc + k1, aN);
    }
#else
    uint4 t0 = {}, t1 = {}, t2 = {}, t3 = {}, ta = {};
    if (hasNext) {
      const uint4* bs = (const uint4*)(wSrc + k1);
      t0 = bs[0]; t1 = bs[1]; t2 = bs[2]; t3 = bs[3];
      ta = *(const uint4*)(aSrc + k1);
      if (kk + 2 < ksteps) __builtin_prefetch(wSrc + k1 + 32, 0, 1);
    }
#endif

    // ---- compute from buffer `cur` ----
    v16h afrag[2];
#pragma unroll
    for (int s = 0; s < 2; ++s) {
      unsigned int* ap = (unsigned int*)&afrag[s];
      const _Float16* abase = &Atile[cur][(wm * 32 + s * 16 + nl) * TPAD];
#pragma unroll
      for (int j = 0; j < 8; ++j) {
        const int kofs = (j << 1) + (h << 4);
        ap[j] = *(const unsigned int*)(abase + kofs);
      }
    }
#pragma unroll
    for (int t = 0; t < 4; ++t) {
      v16h bfrag;
      uint4* bp = (uint4*)&bfrag;
      const _Float16* bbase = &Btile[cur][(wn * 64 + t * 16 + nl) * TPAD + h * 16];
      bp[0] = *(const uint4*)(bbase);
      bp[1] = *(const uint4*)(bbase + 8);
      acc[0][t] = __builtin_amdgcn_wmma_f32_16x16x32_f16(
          false, afrag[0], false, bfrag, (short)0, acc[0][t], false, false);
      acc[1][t] = __builtin_amdgcn_wmma_f32_16x16x32_f16(
          false, afrag[1], false, bfrag, (short)0, acc[1][t], false, false);
    }

#if USE_ASYNC_LDS
    if (hasNext) WAIT_ASYNC();
#else
    if (hasNext) {
      uint4* bd = (uint4*)bN;
      bd[0] = t0; bd[1] = t1; bd[2] = t2; bd[3] = t3;
      *(uint4*)aN = ta;
    }
#endif
    __syncthreads();
  }

  // ---- epilogue: lane holds out[m = j + 8h][n = nl] per subtile ----
#pragma unroll
  for (int s = 0; s < 2; ++s) {
#pragma unroll
    for (int t = 0; t < 4; ++t) {
      const int col = nBase + wn * 64 + t * 16 + nl;
      const float bv = bias ? bias[col] : 0.0f;
#pragma unroll
      for (int j = 0; j < 8; ++j) {
        const int row = mBase + wm * 32 + s * 16 + h * 8 + j;
        float v = acc[s][t][j] + bv;
        if (act == 1) v = gelu_exact(v);
        if (outF32) outF32[(size_t)row * Nout + col] = v;
        if (outF16) outF16[(size_t)row * Nout + col] = (_Float16)v;
      }
    }
  }
}

// ---- pre-attention LayerNorm over all L=258 tokens ----------------------
__launch_bounds__(256)
__global__ void ln_pre_kernel(const float* __restrict__ pre_cls,
                              const float* __restrict__ cls_t,
                              const float* __restrict__ cls_p,
                              const float* __restrict__ g,
                              const float* __restrict__ beta,
                              _Float16* __restrict__ normed) {
  __shared__ float red[256];
  const int row = blockIdx.x;                  // 0 .. B*L-1
  const int b = row / L_;
  const int l = row - b * L_;
  const float* src = (l == 0) ? cls_t : (l == 1) ? cls_p
                   : pre_cls + ((size_t)b * N_ + (l - 2)) * D_;
  const int tid = threadIdx.x;
  const float x0 = src[tid], x1 = src[tid + 256];
  red[tid] = x0 + x1; __syncthreads();
  for (int s = 128; s > 0; s >>= 1) { if (tid < s) red[tid] += red[tid + s]; __syncthreads(); }
  const float mean = red[0] * (1.0f / D_);
  __syncthreads();
  const float d0 = x0 - mean, d1 = x1 - mean;
  red[tid] = d0 * d0 + d1 * d1; __syncthreads();
  for (int s = 128; s > 0; s >>= 1) { if (tid < s) red[tid] += red[tid + s]; __syncthreads(); }
  const float inv = rsqrtf(red[0] * (1.0f / D_) + 1e-5f);
  _Float16* dst = normed + (size_t)row * D_;
  dst[tid]       = (_Float16)(d0 * inv * g[tid] + beta[tid]);
  dst[tid + 256] = (_Float16)(d1 * inv * g[tid + 256] + beta[tid + 256]);
}

// gather normed rows 0,1 of each batch -> compact Q input (512 x 512)
__global__ void gather_q_kernel(const _Float16* __restrict__ normed,
                                _Float16* __restrict__ qin) {
  const int i = blockIdx.x * blockDim.x + threadIdx.x;   // 512*512
  const int r = i >> 9, c = i & 511;                     // r = b*2+t
  const int b = r >> 1, t = r & 1;
  qin[i] = normed[((size_t)b * L_ + t) * D_ + c];
}

// ---- attention for query rows {0,1} per (b,h); entropy from row 0 -------
#define LPAD 272
__launch_bounds__(256)
__global__ void attn_kernel(const float* __restrict__ q32,
                            const _Float16* __restrict__ k16,
                            const _Float16* __restrict__ v16,
                            const unsigned char* __restrict__ mask,
                            _Float16* __restrict__ attnrow16,
                            float* __restrict__ ent_accum) {
  const int hIdx = blockIdx.x;
  const int b    = blockIdx.y;
  const int tid  = threadIdx.x;
  __shared__ float qs[2 * DH_];
  __shared__ float s0[LPAD];
  __shared__ float s1[LPAD];
  __shared__ float red[256];

  if (tid < 2 * DH_) {
    const int t = tid >> 6, d = tid & 63;
    qs[tid] = q32[(size_t)(b * 2 + t) * D_ + hIdx * DH_ + d];
  }
  __syncthreads();

  for (int j = tid; j < LPAD; j += 256) {
    float v0 = -__builtin_inff(), v1 = -__builtin_inff();
    if (j < L_) {
      const bool valid = (j < 2) || (mask[b * N_ + (j - 2)] != 0);
      if (valid) {
        const _Float16* kp = k16 + ((size_t)b * L_ + j) * D_ + hIdx * DH_;
        float d0 = 0.f, d1 = 0.f;
#pragma unroll 8
        for (int d = 0; d < DH_; ++d) {
          const float kv = (float)kp[d];
          d0 += qs[d] * kv;
          d1 += qs[DH_ + d] * kv;
        }
        v0 = d0 * 0.125f; v1 = d1 * 0.125f;   // 1/sqrt(64)
      }
    }
    s0[j] = v0; s1[j] = v1;
  }
  __syncthreads();

  float lm0 = -__builtin_inff(), lm1 = -__builtin_inff();
  for (int j = tid; j < L_; j += 256) { lm0 = fmaxf(lm0, s0[j]); lm1 = fmaxf(lm1, s1[j]); }
  red[tid] = lm0; __syncthreads();
  for (int s = 128; s > 0; s >>= 1) { if (tid < s) red[tid] = fmaxf(red[tid], red[tid + s]); __syncthreads(); }
  const float m0 = red[0]; __syncthreads();
  red[tid] = lm1; __syncthreads();
  for (int s = 128; s > 0; s >>= 1) { if (tid < s) red[tid] = fmaxf(red[tid], red[tid + s]); __syncthreads(); }
  const float m1 = red[0]; __syncthreads();

  float ls0 = 0.f, ls1 = 0.f;
  for (int j = tid; j < LPAD; j += 256) {
    const float e0 = (j < L_) ? expf(s0[j] - m0) : 0.f;
    const float e1 = (j < L_) ? expf(s1[j] - m1) : 0.f;
    s0[j] = e0; s1[j] = e1; ls0 += e0; ls1 += e1;
  }
  __syncthreads();
  red[tid] = ls0; __syncthreads();
  for (int s = 128; s > 0; s >>= 1) { if (tid < s) red[tid] += red[tid + s]; __syncthreads(); }
  const float inv0 = 1.0f / red[0]; __syncthreads();
  red[tid] = ls1; __syncthreads();
  for (int s = 128; s > 0; s >>= 1) { if (tid < s) red[tid] += red[tid + s]; __syncthreads(); }
  const float inv1 = 1.0f / red[0]; __syncthreads();

  float lent = 0.f;
  for (int j = tid; j < L_; j += 256) {
    const float w0 = s0[j] * inv0;
    const float w1 = s1[j] * inv1;
    s0[j] = w0; s1[j] = w1;
    lent += -w0 * logf(w0 + 1e-8f);
  }
  __syncthreads();
  red[tid] = lent; __syncthreads();
  for (int s = 128; s > 0; s >>= 1) { if (tid < s) red[tid] += red[tid + s]; __syncthreads(); }
  if (tid == 0) atomicAdd(ent_accum, red[0]);
  __syncthreads();

  if (tid < 128) {
    const int t = tid >> 6, d = tid & 63;
    const float* w = t ? s1 : s0;
    const _Float16* vp = v16 + (size_t)b * L_ * D_ + hIdx * DH_ + d;
    float acc = 0.f;
    for (int j = 0; j < L_; ++j) acc += w[j] * (float)vp[(size_t)j * D_];
    attnrow16[(size_t)(b * 2 + t) * D_ + hIdx * DH_ + d] = (_Float16)acc;
  }
}

// ---- post-attention LayerNorm on cls rows (residual with cls_t/cls_p) ---
__launch_bounds__(256)
__global__ void post_ln_kernel(const float* __restrict__ proj,
                               const float* __restrict__ cls_t,
                               const float* __restrict__ cls_p,
                               const float* __restrict__ g,
                               const float* __restrict__ beta,
                               float* __restrict__ cls_out) {
  __shared__ float red[256];
  const int row = blockIdx.x;       // b*2 + t
  const int t = row & 1;
  const float* base = t ? cls_p : cls_t;
  const float* pr = proj + (size_t)row * D_;
  const int tid = threadIdx.x;
  const float x0 = base[tid] + pr[tid];
  const float x1 = base[tid + 256] + pr[tid + 256];
  red[tid] = x0 + x1; __syncthreads();
  for (int s = 128; s > 0; s >>= 1) { if (tid < s) red[tid] += red[tid + s]; __syncthreads(); }
  const float mean = red[0] * (1.0f / D_);
  __syncthreads();
  const float d0 = x0 - mean, d1 = x1 - mean;
  red[tid] = d0 * d0 + d1 * d1; __syncthreads();
  for (int s = 128; s > 0; s >>= 1) { if (tid < s) red[tid] += red[tid + s]; __syncthreads(); }
  const float inv = rsqrtf(red[0] * (1.0f / D_) + 1e-5f);
  float* dst = cls_out + (size_t)row * D_;
  dst[tid]       = d0 * inv * g[tid] + beta[tid];
  dst[tid + 256] = d1 * inv * g[tid + 256] + beta[tid + 256];
}

// ---- gate: gv = sigmoid(hs@gate_w^T+gate_b) * (hs@hp_w^T+hp_b) ----------
__launch_bounds__(256)
__global__ void gate_kernel(const float* __restrict__ hav_sum,
                            const float* __restrict__ hav_max,
                            const float* __restrict__ hav_mean,
                            const float* __restrict__ dist_hist,
                            const float* __restrict__ pair_dists,
                            const float* __restrict__ gate_w,
                            const float* __restrict__ gate_b,
                            const float* __restrict__ hp_w,
                            const float* __restrict__ hp_b,
                            const float* __restrict__ cls32,
                            _Float16* __restrict__ pt_in,
                            _Float16* __restrict__ pp_in) {
  const int i = blockIdx.x * blockDim.x + threadIdx.x;   // B*D
  const int b = i >> 9, d = i & 511;
  float hs[GEO_];
  hs[0] = log1pf(hav_sum[b]);
  hs[1] = log1pf(hav_max[b]);
  hs[2] = log1pf(hav_mean[b]);
#pragma unroll
  for (int k = 0; k < NB_; ++k) hs[3 + k] = dist_hist[b * NB_ + k];
#pragma unroll
  for (int k = 0; k < NP_; ++k) hs[3 + NB_ + k] = pair_dists[b * NP_ + k];
  const float* gw = gate_w + (size_t)d * GEO_;
  const float* hw = hp_w + (size_t)d * GEO_;
  float z = gate_b[d], y = hp_b[d];
#pragma unroll
  for (int k = 0; k < GEO_; ++k) { z += hs[k] * gw[k]; y += hs[k] * hw[k]; }
  const float gv = (1.0f / (1.0f + expf(-z))) * y;
  pt_in[i] = (_Float16)(cls32[((size_t)b * 2 + 0) * D_ + d] + gv);
  pp_in[i] = (_Float16)(cls32[((size_t)b * 2 + 1) * D_ + d] + gv);
}

// ---- finalize: entropy scaling + all-masked override --------------------
__launch_bounds__(256)
__global__ void finalize_kernel(const unsigned char* __restrict__ mask,
                                const float* __restrict__ ent_accum,
                                const float* __restrict__ miss_t,
                                const float* __restrict__ miss_p,
                                float* __restrict__ out) {
  __shared__ int anyv[256];
  __shared__ int cnt[256];
  const int b = threadIdx.x;
  int any = 0;
  for (int j = 0; j < N_; ++j) any |= (mask[b * N_ + j] != 0);
  anyv[b] = any; cnt[b] = any; __syncthreads();
  for (int s = 128; s > 0; s >>= 1) { if (b < s) cnt[b] += cnt[b + s]; __syncthreads(); }
  if (b == 0) {
    const float factor = (float)cnt[0] / (float)B_;
    out[2 * B_ * D_] = (ent_accum[0] / (float)(B_ * H_)) * factor;
  }
  if (!anyv[b]) {
    for (int d = 0; d < D_; ++d) {
      out[(size_t)b * D_ + d] = miss_t[d];
      out[(size_t)B_ * D_ + (size_t)b * D_ + d] = miss_p[d];
    }
  }
}

// ---- host orchestration -------------------------------------------------
extern "C" void kernel_launch(void* const* d_in, const int* in_sizes, int n_in,
                              void* d_out, int out_size, void* d_ws, size_t ws_size,
                              hipStream_t stream) {
  (void)in_sizes; (void)n_in; (void)out_size; (void)ws_size;

  const int*           step_ids  = (const int*)d_in[0];
  const float*         coords    = (const float*)d_in[1];
  const unsigned char* mask      = (const unsigned char*)d_in[2];   // bool8
  const float* hav_sum   = (const float*)d_in[3];
  const float* hav_max   = (const float*)d_in[4];
  const float* hav_mean  = (const float*)d_in[5];
  const float* dist_hist = (const float*)d_in[6];
  const float* pair_dists= (const float*)d_in[7];
  const float* emb       = (const float*)d_in[8];
  const float* token_w   = (const float*)d_in[9];
  const float* token_b   = (const float*)d_in[10];
  const float* cls_t     = (const float*)d_in[11];
  const float* cls_p     = (const float*)d_in[12];
  const float* pre_g     = (const float*)d_in[13];
  const float* pre_b     = (const float*)d_in[14];
  const float* wq        = (const float*)d_in[15];
  const float* wk        = (const float*)d_in[16];
  const float* wv        = (const float*)d_in[17];
  const float* bq        = (const float*)d_in[18];
  const float* bk        = (const float*)d_in[19];
  const float* bv        = (const float*)d_in[20];
  const float* wo        = (const float*)d_in[21];
  const float* bo        = (const float*)d_in[22];
  const float* post_g    = (const float*)d_in[23];
  const float* post_b    = (const float*)d_in[24];
  const float* gate_w    = (const float*)d_in[25];
  const float* gate_b    = (const float*)d_in[26];
  const float* hp_w      = (const float*)d_in[27];
  const float* hp_b      = (const float*)d_in[28];
  const float* tp_w      = (const float*)d_in[29];
  const float* tp_b      = (const float*)d_in[30];
  const float* pp_w      = (const float*)d_in[31];
  const float* pp_b      = (const float*)d_in[32];
  const float* miss_t    = (const float*)d_in[33];
  const float* miss_p    = (const float*)d_in[34];

  float* out = (float*)d_out;
  float* out_pt     = out;                             // (B, D)
  float* out_pp     = out + (size_t)B_ * D_;           // (B, D)
  float* out_pre    = out + 2 * (size_t)B_ * D_ + 1;   // (B, N, D)

  // workspace layout (256B aligned)
  char* ws = (char*)d_ws;
  size_t off = 0;
  auto take = [&](size_t bytes) -> char* {
    char* p = ws + off;
    off = (off + bytes + 255) & ~(size_t)255;
    return p;
  };
  const size_t BL = (size_t)B_ * L_;
  _Float16* normed16 = (_Float16*)take(BL * D_ * 2);          // 67.6 MB
  _Float16* k16      = (_Float16*)take(BL * D_ * 2);          // 67.6 MB
  _Float16* v16      = (_Float16*)take(BL * D_ * 2);          // 67.6 MB
  _Float16* w16tok   = (_Float16*)take((size_t)D_ * (E_ + CD_) * 2);
  _Float16* w16q     = (_Float16*)take((size_t)D_ * D_ * 2);
  _Float16* w16k     = (_Float16*)take((size_t)D_ * D_ * 2);
  _Float16* w16v     = (_Float16*)take((size_t)D_ * D_ * 2);
  _Float16* w16o     = (_Float16*)take((size_t)D_ * D_ * 2);
  _Float16* w16tp    = (_Float16*)take((size_t)D_ * D_ * 2);
  _Float16* w16pp    = (_Float16*)take((size_t)D_ * D_ * 2);
  _Float16* qin16    = (_Float16*)take((size_t)2 * B_ * D_ * 2);
  float*    q32      = (float*)take((size_t)2 * B_ * D_ * 4);
  _Float16* attn16   = (_Float16*)take((size_t)2 * B_ * D_ * 2);
  float*    proj32   = (float*)take((size_t)2 * B_ * D_ * 4);
  float*    cls32    = (float*)take((size_t)2 * B_ * D_ * 4);
  _Float16* pt_in16  = (_Float16*)take((size_t)B_ * D_ * 2);
  _Float16* pp_in16  = (_Float16*)take((size_t)B_ * D_ * 2);
  float*    ent_ws   = (float*)take(256);
  // xin16 (B*N x 320 f16, 41.9 MB) aliases k16: dead before k16 is written
  _Float16* xin16 = k16;

  zero_scalar_kernel<<<1, 1, 0, stream>>>(ent_ws);

  // weight conversions
  f32_to_f16_kernel<<<(D_ * (E_ + CD_)) / 256, 256, 0, stream>>>(token_w, w16tok, D_ * (E_ + CD_));
  f32_to_f16_kernel<<<(D_ * D_) / 256, 256, 0, stream>>>(wq, w16q, D_ * D_);
  f32_to_f16_kernel<<<(D_ * D_) / 256, 256, 0, stream>>>(wk, w16k, D_ * D_);
  f32_to_f16_kernel<<<(D_ * D_) / 256, 256, 0, stream>>>(wv, w16v, D_ * D_);
  f32_to_f16_kernel<<<(D_ * D_) / 256, 256, 0, stream>>>(wo, w16o, D_ * D_);
  f32_to_f16_kernel<<<(D_ * D_) / 256, 256, 0, stream>>>(tp_w, w16tp, D_ * D_);
  f32_to_f16_kernel<<<(D_ * D_) / 256, 256, 0, stream>>>(pp_w, w16pp, D_ * D_);

  // embed + concat -> f16 activation matrix (65536 x 320)
  build_xin_kernel<<<(B_ * N_ * (E_ + CD_)) / 256, 256, 0, stream>>>(
      step_ids, coords, emb, xin16);

  // token GEMM + GELU -> pre_cls (f32, straight into d_out)
  wmma_gemm_kernel<<<dim3((B_ * N_) / GT_M, D_ / GT_N), 256, 0, stream>>>(
      xin16, w16tok, token_b, out_pre, nullptr, B_ * N_, E_ + CD_, D_, 1);

  // pre-LN over all tokens -> normed16
  ln_pre_kernel<<<B_ * L_, 256, 0, stream>>>(out_pre, cls_t, cls_p, pre_g, pre_b, normed16);

  // K, V projections (f16 outputs)
  wmma_gemm_kernel<<<dim3((int)(BL / GT_M), D_ / GT_N), 256, 0, stream>>>(
      normed16, w16k, bk, nullptr, k16, (int)BL, D_, D_, 0);
  wmma_gemm_kernel<<<dim3((int)(BL / GT_M), D_ / GT_N), 256, 0, stream>>>(
      normed16, w16v, bv, nullptr, v16, (int)BL, D_, D_, 0);

  // Q projection only for cls rows (512 rows)
  gather_q_kernel<<<(2 * B_ * D_) / 256, 256, 0, stream>>>(normed16, qin16);
  wmma_gemm_kernel<<<dim3((2 * B_) / GT_M, D_ / GT_N), 256, 0, stream>>>(
      qin16, w16q, bq, q32, nullptr, 2 * B_, D_, D_, 0);

  // attention (2 query rows per batch) + entropy accumulation
  attn_kernel<<<dim3(H_, B_), 256, 0, stream>>>(q32, k16, v16, mask, attn16, ent_ws);

  // output projection on cls rows
  wmma_gemm_kernel<<<dim3((2 * B_) / GT_M, D_ / GT_N), 256, 0, stream>>>(
      attn16, w16o, bo, proj32, nullptr, 2 * B_, D_, D_, 0);

  // residual + post-LN -> cls32
  post_ln_kernel<<<2 * B_, 256, 0, stream>>>(proj32, cls_t, cls_p, post_g, post_b, cls32);

  // gate + gv; build pt/pp GEMM inputs
  gate_kernel<<<(B_ * D_) / 256, 256, 0, stream>>>(
      hav_sum, hav_max, hav_mean, dist_hist, pair_dists,
      gate_w, gate_b, hp_w, hp_b, cls32, pt_in16, pp_in16);

  // pt / pp GEMMs with GELU, straight into d_out
  wmma_gemm_kernel<<<dim3(B_ / GT_M, D_ / GT_N), 256, 0, stream>>>(
      pt_in16, w16tp, tp_b, out_pt, nullptr, B_, D_, D_, 1);
  wmma_gemm_kernel<<<dim3(B_ / GT_M, D_ / GT_N), 256, 0, stream>>>(
      pp_in16, w16pp, pp_b, out_pp, nullptr, B_, D_, D_, 1);

  // entropy scale + all-masked override
  finalize_kernel<<<1, 256, 0, stream>>>(mask, ent_ws, miss_t, miss_p, out);
}
